// MultiHeadAttention_48447231099260
// MI455X (gfx1250) — matrix operand order repaired and checked by reading, compiled, not run on hardware
//
#include <hip/hip_runtime.h>

// ---------------------------------------------------------------------------
// MHA forward for MI455X (gfx1250): bf16 WMMA, f32 accumulation, flash
// attention with TRANSPOSED score tiles (S^T = K.Q^T) so the online softmax
// is in-lane and P needs no re-layout for the P.V WMMA.  K tiles are staged
// to LDS with gfx1250 async copies (ASYNCcnt) when available; next tiles are
// prefetched with global_prefetch.
// D_MODEL=768, H=12, d_k=64, B=2, S=4096.
// ---------------------------------------------------------------------------

typedef __bf16 bf16_t;
typedef __attribute__((ext_vector_type(16))) __bf16 v16bf;
typedef __attribute__((ext_vector_type(8)))  __bf16 v8bf;
typedef __attribute__((ext_vector_type(8)))  float  v8f;

#define D_MODEL 768
#define SEQ     4096
#define BATCH   2
#define HEADS   12
#define DK      64
#define ROWS    (BATCH * SEQ)          // 8192

// Async global->LDS copy (device pass only; host pass sees fallback).
#if defined(__gfx1250__) && __has_builtin(__builtin_amdgcn_global_load_async_to_lds_b128)
#define HAVE_ASYNC_LDS 1
#else
#define HAVE_ASYNC_LDS 0
#endif

#if HAVE_ASYNC_LDS
typedef int v4i_t __attribute__((ext_vector_type(4)));
typedef __attribute__((address_space(1))) v4i_t* gptr_v4i;
typedef __attribute__((address_space(3))) v4i_t* lptr_v4i;
static __device__ inline void async_copy_b128(const void* g, void* l) {
    __builtin_amdgcn_global_load_async_to_lds_b128((gptr_v4i)g, (lptr_v4i)l, 0, 0);
}
static __device__ inline void wait_async0() {
#if __has_builtin(__builtin_amdgcn_s_wait_asynccnt)
    __builtin_amdgcn_s_wait_asynccnt(0);
#else
    asm volatile("s_wait_asynccnt 0x0" ::: "memory");
#endif
}
#endif

static __device__ inline v8f vzero8() {
    v8f z;
#pragma unroll
    for (int i = 0; i < 8; ++i) z[i] = 0.0f;
    return z;
}

static __device__ inline v8f wmma_bf16(v16bf a, v16bf b, v8f c) {
    return __builtin_amdgcn_wmma_f32_16x16x32_bf16(
        /*neg_a=*/false, a, /*neg_b=*/false, b,
        /*c_mod=*/(short)0, c, /*reuse_a=*/false, /*reuse_b=*/false);
}

// A-matrix fragment (16x32 bf16, row-major source, ld in elements).
// lane L: row = row0 + (L&15); elems 0..7 -> k0+kb..+7, elems 8..15 -> k0+kb+16..+23
static __device__ inline v16bf load_afrag(const bf16_t* __restrict__ base,
                                          int ld, int row0, int k0, int lane) {
    const int r  = row0 + (lane & 15);
    const int kb = (lane >> 4) * 8;
    const bf16_t* p = base + (size_t)r * ld + k0 + kb;
    v8bf lo = *(const v8bf*)(p);
    v8bf hi = *(const v8bf*)(p + 16);
    v16bf a;
#pragma unroll
    for (int j = 0; j < 8; ++j) { a[j] = lo[j]; a[j + 8] = hi[j]; }
    return a;
}

// B-matrix fragment where B[k][n] = Src[n][k] (Src row-major, ld elements):
// lane L: col = col0 + (L&15); elems 0..7 -> k0+kb..+7, elems 8..15 -> k0+kb+16..+23
static __device__ inline v16bf load_bfrag_rowmajorT(const bf16_t* __restrict__ S,
                                                    int ld, int col0, int k0, int lane) {
    const int c   = col0 + (lane & 15);
    const int kb  = (lane >> 4) * 8;
    const bf16_t* p = S + (size_t)c * ld + k0 + kb;
    v8bf lo = *(const v8bf*)(p);
    v8bf hi = *(const v8bf*)(p + 16);
    v16bf r;
#pragma unroll
    for (int j = 0; j < 8; ++j) { r[j] = lo[j]; r[j + 8] = hi[j]; }
    return r;
}

// ---------------------------------------------------------------------------
// Elementwise cast f32 -> bf16
// ---------------------------------------------------------------------------
__global__ void cast_f32_bf16(const float* __restrict__ in,
                              bf16_t* __restrict__ out, int n) {
    int i = (blockIdx.x * blockDim.x + threadIdx.x) * 4;
    if (i + 3 < n) {
        float4 v = *(const float4*)(in + i);
        out[i + 0] = (bf16_t)v.x;
        out[i + 1] = (bf16_t)v.y;
        out[i + 2] = (bf16_t)v.z;
        out[i + 3] = (bf16_t)v.w;
    }
}

// ---------------------------------------------------------------------------
// GEMM: C[M,N] = (A[M,K] * W[N,K]^T + bias[N]) * oscale.
// One wave -> 32x64 tile (8 WMMAs per k-step).
// grid = (M/32)*(N/64)/8 blocks of 256 threads.
// ---------------------------------------------------------------------------
template <bool OUT_BF16>
__global__ __launch_bounds__(256)
void gemm_xwt(const bf16_t* __restrict__ A, const bf16_t* __restrict__ W,
              const float* __restrict__ bias,
              bf16_t* __restrict__ Cb, float* __restrict__ Cf,
              int M, int N, int K, float oscale) {
    const int lane   = threadIdx.x & 31;
    const int wave   = threadIdx.x >> 5;
    const int mTiles = M >> 5;                 // 32-row tiles
    const int w      = blockIdx.x * 8 + wave;
    const int m0     = (w % mTiles) * 32;
    const int n0     = (w / mTiles) * 64;

    v8f acc[2][4];
#pragma unroll
    for (int r = 0; r < 2; ++r)
#pragma unroll
        for (int t = 0; t < 4; ++t) acc[r][t] = vzero8();

    for (int k0 = 0; k0 < K; k0 += 32) {
        v16bf a0 = load_afrag(A, K, m0,      k0, lane);
        v16bf a1 = load_afrag(A, K, m0 + 16, k0, lane);
#pragma unroll
        for (int t = 0; t < 4; ++t) {
            v16bf b = load_bfrag_rowmajorT(W, K, n0 + t * 16, k0, lane);
            acc[0][t] = wmma_bf16(a0, b, acc[0][t]);
            acc[1][t] = wmma_bf16(a1, b, acc[1][t]);
        }
    }

#pragma unroll
    for (int r = 0; r < 2; ++r) {
        const int rbase = m0 + r * 16 + (lane >> 4) * 8;
#pragma unroll
        for (int t = 0; t < 4; ++t) {
            const int col = n0 + t * 16 + (lane & 15);
            const float bv = bias[col];
#pragma unroll
            for (int i = 0; i < 8; ++i) {
                const float v = (acc[r][t][i] + bv) * oscale;
                const size_t idx = (size_t)(rbase + i) * N + col;
                if (OUT_BF16) Cb[idx] = (bf16_t)v;
                else          Cf[idx] = v;
            }
        }
    }
}

// ---------------------------------------------------------------------------
// Flash attention with transposed score tiles.
// Q/K/V stored as [B*S, 768] bf16 with head h at cols h*64..h*64+63.
// Q is pre-scaled by 1/sqrt(d_k) (folded into the Q projection).
// Block = 8 waves = 128 queries of one (b,h); key tiles of 32.
// grid.x = B*H * (S/128) = 768 blocks.
// ---------------------------------------------------------------------------
__global__ __launch_bounds__(256)
void flash_attn(const bf16_t* __restrict__ Qg, const bf16_t* __restrict__ Kg,
                const bf16_t* __restrict__ Vg, bf16_t* __restrict__ Og) {
    __shared__ __align__(64) bf16_t Kt[32][64];      // [key][d]
    __shared__ __align__(64) bf16_t Vt[64][32];      // [d][key] (transposed)

    const int lane = threadIdx.x & 31;
    const int wave = threadIdx.x >> 5;
    const int bh   = blockIdx.x >> 5;                // 0..23
    const int qBlk = blockIdx.x & 31;                // 0..31
    const int b    = bh / HEADS;
    const int h    = bh % HEADS;
    const size_t base = (size_t)b * SEQ * D_MODEL + (size_t)h * DK;
    const int q0   = qBlk * 128 + wave * 16;

    // Q as B-fragments of S^T = K.Q^T : B[k=d][n=query] = Q[query][d].
    const v16bf qb0 = load_bfrag_rowmajorT(Qg + base, D_MODEL, q0, 0,  lane);
    const v16bf qb1 = load_bfrag_rowmajorT(Qg + base, D_MODEL, q0, 32, lane);

    // Per-lane softmax state for query (lane&15) (replicated in both halves).
    float mq = -__builtin_inff();
    float lq = 0.0f;
    v8f o[4];
#pragma unroll
    for (int t = 0; t < 4; ++t) o[t] = vzero8();

    const int nn  = lane & 15;
    const int kb2 = (lane >> 4) * 16;
    const int rowSel = 8 * (lane >> 4);              // O-row selector base

    // Per-thread staging coordinates (16B chunk of a 32x64 bf16 tile).
    const int sr = threadIdx.x >> 3;                 // 0..31 key row
    const int sc = (threadIdx.x & 7) * 8;            // 0..56 d chunk

    for (int kt = 0; kt < SEQ; kt += 32) {
        __syncthreads();   // previous tile fully consumed before overwrite
        {
            const size_t g = base + (size_t)(kt + sr) * D_MODEL + sc;
#if HAVE_ASYNC_LDS
            async_copy_b128(Kg + g, &Kt[sr][sc]);    // DMA straight to LDS
#else
            *(v8bf*)&Kt[sr][sc] = *(const v8bf*)(Kg + g);
#endif
            v8bf vv = *(const v8bf*)(Vg + g);        // V needs a transpose
#pragma unroll
            for (int j = 0; j < 8; ++j) Vt[sc + j][sr] = vv[j];

            if (kt + 32 < SEQ) {                     // warm L2 for next tile
                const size_t gn = g + (size_t)32 * D_MODEL;
                __builtin_prefetch((const void*)(Kg + gn), 0, 3);
                __builtin_prefetch((const void*)(Vg + gn), 0, 3);
            }
#if HAVE_ASYNC_LDS
            wait_async0();
#endif
        }
        __syncthreads();

        // ---- S^T tiles: A = K rows (from LDS), B = Q^T (in regs) ----
        v8f s0 = vzero8(), s1 = vzero8();            // keys 0-15 / 16-31
        {
            v16bf ka0 = load_afrag(&Kt[0][0], 64, 0,  0,  lane);
            v16bf ka1 = load_afrag(&Kt[0][0], 64, 0,  32, lane);
            v16bf kb0 = load_afrag(&Kt[0][0], 64, 16, 0,  lane);
            v16bf kb1 = load_afrag(&Kt[0][0], 64, 16, 32, lane);
            s0 = wmma_bf16(ka0, qb0, s0);
            s0 = wmma_bf16(ka1, qb1, s0);
            s1 = wmma_bf16(kb0, qb0, s1);
            s1 = wmma_bf16(kb1, qb1, s1);
        }

        // ---- online softmax: keys live in-register, query = lane column ----
        float lm = fmaxf(s0[0], s1[0]);
#pragma unroll
        for (int i = 1; i < 8; ++i) lm = fmaxf(lm, fmaxf(s0[i], s1[i]));
        lm = fmaxf(lm, __shfl_xor(lm, 16));          // combine lane halves
        const float mnew  = fmaxf(mq, lm);
        const float alpha = __expf(mq - mnew);
        float e0[8], e1[8], ls = 0.0f;
#pragma unroll
        for (int i = 0; i < 8; ++i) {
            e0[i] = __expf(s0[i] - mnew);
            e1[i] = __expf(s1[i] - mnew);
            ls += e0[i] + e1[i];
        }
        ls += __shfl_xor(ls, 16);
        lq = lq * alpha + ls;
        mq = mnew;

        // ---- P^T C-layout == P A-layout: in-lane pack to bf16 ----
        v16bf pf;
#pragma unroll
        for (int j = 0; j < 8; ++j) {
            pf[j]     = (bf16_t)e0[j];
            pf[j + 8] = (bf16_t)e1[j];
        }

        // ---- rescale O by per-query alpha (broadcast into O row layout) ----
#pragma unroll
        for (int i = 0; i < 8; ++i) {
            const float ar = __shfl(alpha, i + rowSel);
            o[0][i] *= ar; o[1][i] *= ar; o[2][i] *= ar; o[3][i] *= ar;
        }

        // ---- O += P . V  (V^T staged so B-frags are contiguous) ----
#pragma unroll
        for (int t = 0; t < 4; ++t) {
            v8bf vlo = *(const v8bf*)&Vt[t * 16 + nn][kb2 / 2];
            v8bf vhi = *(const v8bf*)&Vt[t * 16 + nn][kb2 / 2 + 16];
            v16bf bv;
#pragma unroll
            for (int j = 0; j < 8; ++j) { bv[j] = vlo[j]; bv[j + 8] = vhi[j]; }
            o[t] = wmma_bf16(pf, bv, o[t]);
        }
    }

    // ---- epilogue: normalize by 1/l (broadcast) and store bf16 ----
    const float linv = 1.0f / lq;
    const int rbase = q0 + rowSel;
#pragma unroll
    for (int i = 0; i < 8; ++i) {
        const float nr = __shfl(linv, i + rowSel);
#pragma unroll
        for (int t = 0; t < 4; ++t) {
            const int d = t * 16 + nn;
            Og[base + (size_t)(rbase + i) * D_MODEL + d] = (bf16_t)(o[t][i] * nr);
        }
    }
}

// ---------------------------------------------------------------------------
// Host-side orchestration
// ---------------------------------------------------------------------------
extern "C" void kernel_launch(void* const* d_in, const int* in_sizes, int n_in,
                              void* d_out, int out_size, void* d_ws, size_t ws_size,
                              hipStream_t stream) {
    const float* x  = (const float*)d_in[0];
    const float* Wq = (const float*)d_in[1];
    const float* bq = (const float*)d_in[2];
    const float* Wk = (const float*)d_in[3];
    const float* bk = (const float*)d_in[4];
    const float* Wv = (const float*)d_in[5];
    const float* bv = (const float*)d_in[6];
    const float* Wo = (const float*)d_in[7];
    const float* bo = (const float*)d_in[8];
    float* out = (float*)d_out;

    // carve workspace (256B aligned slabs)
    char* ws = (char*)d_ws;
    size_t off = 0;
    auto alloc = [&](size_t bytes) {
        char* p = ws + off;
        off = (off + bytes + 255) & ~(size_t)255;
        return p;
    };
    const size_t xElems = (size_t)ROWS * D_MODEL;        // 6,291,456
    const size_t wElems = (size_t)D_MODEL * D_MODEL;     //   589,824
    bf16_t* Xb  = (bf16_t*)alloc(xElems * 2);
    bf16_t* Wqb = (bf16_t*)alloc(wElems * 2);
    bf16_t* Wkb = (bf16_t*)alloc(wElems * 2);
    bf16_t* Wvb = (bf16_t*)alloc(wElems * 2);
    bf16_t* Wob = (bf16_t*)alloc(wElems * 2);
    bf16_t* Qb  = (bf16_t*)alloc(xElems * 2);
    bf16_t* Kb  = (bf16_t*)alloc(xElems * 2);
    bf16_t* Vb  = (bf16_t*)alloc(xElems * 2);
    bf16_t* AOb = (bf16_t*)alloc(xElems * 2);
    (void)ws_size; (void)n_in; (void)in_sizes; (void)out_size;

    // 1) casts to bf16
    {
        int n = (int)xElems;
        cast_f32_bf16<<<(n / 4 + 255) / 256, 256, 0, stream>>>(x, Xb, n);
        int m = (int)wElems;
        int blocks = (m / 4 + 255) / 256;
        cast_f32_bf16<<<blocks, 256, 0, stream>>>(Wq, Wqb, m);
        cast_f32_bf16<<<blocks, 256, 0, stream>>>(Wk, Wkb, m);
        cast_f32_bf16<<<blocks, 256, 0, stream>>>(Wv, Wvb, m);
        cast_f32_bf16<<<blocks, 256, 0, stream>>>(Wo, Wob, m);
    }

    // 2) Q/K/V projections: (M/32)*(N/64)/8 = 256*12/8 = 384 blocks.
    //    1/sqrt(d_k) folded into the Q projection output.
    const int gemmBlocks = (ROWS / 32) * (D_MODEL / 64) / 8;
    gemm_xwt<true><<<gemmBlocks, 256, 0, stream>>>(Xb, Wqb, bq, Qb, nullptr,
                                                   ROWS, D_MODEL, D_MODEL, 0.125f);
    gemm_xwt<true><<<gemmBlocks, 256, 0, stream>>>(Xb, Wkb, bk, Kb, nullptr,
                                                   ROWS, D_MODEL, D_MODEL, 1.0f);
    gemm_xwt<true><<<gemmBlocks, 256, 0, stream>>>(Xb, Wvb, bv, Vb, nullptr,
                                                   ROWS, D_MODEL, D_MODEL, 1.0f);

    // 3) flash attention: 24 (b,h) * 32 query blocks = 768 blocks
    flash_attn<<<BATCH * HEADS * (SEQ / 128), 256, 0, stream>>>(Qb, Kb, Vb, AOb);

    // 4) output projection -> f32 d_out
    gemm_xwt<false><<<gemmBlocks, 256, 0, stream>>>(AOb, Wob, bo, nullptr, out,
                                                    ROWS, D_MODEL, D_MODEL, 1.0f);
}
